// FusedMoEDispatch_17935783428807
// MI455X (gfx1250) — matrix operand order repaired and checked by reading, not compiled, not run on hardware
//
#include <hip/hip_runtime.h>

#define E_ 8
#define H_ 2048
#define I_ 5632
#define B_ 1024
#define K_ 2

typedef __attribute__((ext_vector_type(16))) __bf16 v16bf;
typedef __attribute__((ext_vector_type(8)))  float  v8f;

union FragAB { v16bf v; uint4 q[2]; };

// scalar f32 -> bf16 (round-to-nearest-even), boundary epilogues only
__device__ __forceinline__ unsigned short f2bf(float f) {
    unsigned int u = __float_as_uint(f);
    unsigned int r = (u + 0x7FFFu + ((u >> 16) & 1u)) >> 16;
    return (unsigned short)r;
}

// hardware packed conversion: dword = { bf16(hi) , bf16(lo) }
__device__ __forceinline__ unsigned pack_bf16(float lo, float hi) {
#if __has_builtin(__builtin_amdgcn_cvt_pk_bf16_f32)
    auto p = __builtin_amdgcn_cvt_pk_bf16_f32(lo, hi);
    unsigned u;
    __builtin_memcpy(&u, &p, 4);
    return u;
#else
    unsigned r;
    asm("v_cvt_pk_bf16_f32 %0, %1, %2" : "=v"(r) : "v"(lo), "v"(hi));
    return r;
#endif
}

// async copy: global (saddr base + 32-bit voff) -> LDS (lds byte offset), 16B
__device__ __forceinline__ void async_ld_b128(unsigned ldsoff, const void* base, int voff) {
    asm volatile("global_load_async_to_lds_b128 %0, %1, %2"
                 : : "v"(ldsoff), "v"(voff), "s"(base) : "memory");
}
__device__ __forceinline__ void wait_async0() {
    asm volatile("s_wait_asynccnt 0x0" : : : "memory");
}
__device__ __forceinline__ unsigned lds_off(const void* p) {
    return (unsigned)(uintptr_t)p;   // low 32 bits of flat LDS address = LDS offset
}

__device__ __forceinline__ v8f vzero8() {
    v8f z;
#pragma unroll
    for (int j = 0; j < 8; ++j) z[j] = 0.f;
    return z;
}

__device__ __forceinline__ v8f wmma_bf16(const FragAB& a, const FragAB& b, v8f c) {
    return __builtin_amdgcn_wmma_f32_16x16x32_bf16(
        false, a.v, false, b.v, (short)0, c, false, false);
}

__device__ __forceinline__ float silu_fast(float g) {
    return g * __builtin_amdgcn_rcpf(1.f + __expf(-g));
}

// ---------------------------------------------------------------- utilities
__global__ void zero_f32(float* __restrict__ p, int n) {
    int i = blockIdx.x * blockDim.x + threadIdx.x;
    if (i < n) p[i] = 0.f;
}

__global__ void cvt_bf16(const float* __restrict__ x, unsigned* __restrict__ xb2, int n2) {
    int i = blockIdx.x * blockDim.x + threadIdx.x;
    if (i < n2) {
        float2 v = ((const float2*)x)[i];
        xb2[i] = pack_bf16(v.x, v.y);
    }
}

// Build per-expert token lists: estart[E+1], tokidx[B*K], tokw[B*K]
__global__ void moe_route(const int* __restrict__ ids, const float* __restrict__ ew,
                          int* __restrict__ estart, int* __restrict__ tokidx,
                          float* __restrict__ tokw) {
    __shared__ int cnt[E_];
    __shared__ int cur[E_];
    const int tid = threadIdx.x;
    if (tid < E_) cnt[tid] = 0;
    __syncthreads();
    for (int i = tid; i < B_ * K_; i += blockDim.x) atomicAdd(&cnt[ids[i]], 1);
    __syncthreads();
    if (tid == 0) {
        int s = 0;
        for (int e = 0; e < E_; ++e) { estart[e] = s; cur[e] = s; s += cnt[e]; }
        estart[E_] = s;
    }
    __syncthreads();
    for (int i = tid; i < B_ * K_; i += blockDim.x) {
        int e = ids[i];
        int p = atomicAdd(&cur[e], 1);
        tokidx[p] = i / K_;
        tokw[p]   = ew[i];
    }
}

// ---------------------------------------------------------------- GEMM 1
// act[es+m, :] = silu(x[tok] @ Wg[e]) * (x[tok] @ Wu[e]),  bf16 output
// Block tile 64x128, K-step 32, 8 waves each 32x32 (8 WMMA/step).
// Software-pipelined double-buffered LDS: one barrier per K-step.
__global__ __launch_bounds__(256) void moe_gemm1(
    const unsigned short* __restrict__ xb,
    const float* __restrict__ Wg, const float* __restrict__ Wu,
    const int* __restrict__ estart, const int* __restrict__ tokidx,
    unsigned short* __restrict__ act) {
    const int e  = blockIdx.y >> 5;
    const int mt = blockIdx.y & 31;
    const int es = estart[e];
    const int ne = estart[e + 1] - es;
    const int m0 = mt * 64;
    if (m0 >= ne) return;
    const int n0 = blockIdx.x * 128;

    __shared__ __align__(16) unsigned short sA [2][64][40];
    __shared__ __align__(16) unsigned short sBg[2][128][40];  // transposed [n][k]
    __shared__ __align__(16) unsigned short sBu[2][128][40];

    const int tid  = threadIdx.x;
    const int lane = tid & 31;
    const int wv   = tid >> 5;
    const int wrow = wv >> 2;   // 0..1
    const int wcol = wv & 3;    // 0..3
    const int lm   = lane & 15;
    const int kh   = lane >> 4;

    const size_t eoff = (size_t)e * H_ * I_;

    v8f accG[4], accU[4];
#pragma unroll
    for (int i = 0; i < 4; ++i) { accG[i] = vzero8(); accU[i] = vzero8(); }

    // A staging: clamped gather (garbage rows only feed rows we never store)
    const int arow = tid >> 2;           // 0..63
    const int aseg = tid & 3;            // 0..3
    const int ar   = (m0 + arow < ne) ? (m0 + arow) : (ne - 1);
    const int atok = tokidx[es + ar];
    const int abase = atok * (H_ * 2) + aseg * 16;
    const unsigned aldso[2] = { lds_off(&sA[0][arow][aseg * 8]),
                                lds_off(&sA[1][arow][aseg * 8]) };

    int kp[2], c4[2];
#pragma unroll
    for (int v = 0; v < 2; ++v) { int idx = v * 256 + tid; kp[v] = idx >> 5; c4[v] = idx & 31; }

#define COMPUTE1(CUR) do {                                                     \
        FragAB a0, a1, bg0, bg1, bu0, bu1;                                     \
        const int m = wrow * 32 + lm;                                          \
        const int n = wcol * 32 + lm;                                          \
        a0.q[0] = *(const uint4*)(&sA[CUR][m][kh * 16]);                       \
        a0.q[1] = *(const uint4*)(&sA[CUR][m][kh * 16 + 8]);                   \
        a1.q[0] = *(const uint4*)(&sA[CUR][m + 16][kh * 16]);                  \
        a1.q[1] = *(const uint4*)(&sA[CUR][m + 16][kh * 16 + 8]);              \
        bg0.q[0] = *(const uint4*)(&sBg[CUR][n][kh * 16]);                     \
        bg0.q[1] = *(const uint4*)(&sBg[CUR][n][kh * 16 + 8]);                 \
        bg1.q[0] = *(const uint4*)(&sBg[CUR][n + 16][kh * 16]);                \
        bg1.q[1] = *(const uint4*)(&sBg[CUR][n + 16][kh * 16 + 8]);            \
        bu0.q[0] = *(const uint4*)(&sBu[CUR][n][kh * 16]);                     \
        bu0.q[1] = *(const uint4*)(&sBu[CUR][n][kh * 16 + 8]);                 \
        bu1.q[0] = *(const uint4*)(&sBu[CUR][n + 16][kh * 16]);                \
        bu1.q[1] = *(const uint4*)(&sBu[CUR][n + 16][kh * 16 + 8]);            \
        accG[0] = wmma_bf16(a0, bg0, accG[0]);                                 \
        accG[1] = wmma_bf16(a0, bg1, accG[1]);                                 \
        accG[2] = wmma_bf16(a1, bg0, accG[2]);                                 \
        accG[3] = wmma_bf16(a1, bg1, accG[3]);                                 \
        accU[0] = wmma_bf16(a0, bu0, accU[0]);                                 \
        accU[1] = wmma_bf16(a0, bu1, accU[1]);                                 \
        accU[2] = wmma_bf16(a1, bu0, accU[2]);                                 \
        accU[3] = wmma_bf16(a1, bu1, accU[3]);                                 \
    } while (0)

    // prologue: stage k0 = 0 into buffer 0
    {
        async_ld_b128(aldso[0], xb, abase);
#pragma unroll
        for (int v = 0; v < 2; ++v) {
            const size_t o = eoff + (size_t)(2 * kp[v]) * I_ + n0 + c4[v] * 4;
            float4 ga = *(const float4*)(Wg + o), gb = *(const float4*)(Wg + o + I_);
            float4 ua = *(const float4*)(Wu + o), ub = *(const float4*)(Wu + o + I_);
            int nn = c4[v] * 4, kk = 2 * kp[v];
            *(unsigned*)&sBg[0][nn + 0][kk] = pack_bf16(ga.x, gb.x);
            *(unsigned*)&sBg[0][nn + 1][kk] = pack_bf16(ga.y, gb.y);
            *(unsigned*)&sBg[0][nn + 2][kk] = pack_bf16(ga.z, gb.z);
            *(unsigned*)&sBg[0][nn + 3][kk] = pack_bf16(ga.w, gb.w);
            *(unsigned*)&sBu[0][nn + 0][kk] = pack_bf16(ua.x, ub.x);
            *(unsigned*)&sBu[0][nn + 1][kk] = pack_bf16(ua.y, ub.y);
            *(unsigned*)&sBu[0][nn + 2][kk] = pack_bf16(ua.z, ub.z);
            *(unsigned*)&sBu[0][nn + 3][kk] = pack_bf16(ua.w, ub.w);
        }
        wait_async0();
        __syncthreads();
    }

    // strength-reduced element offsets (shared by gate & up) for next tile (kn=32)
    size_t goff[2];
#pragma unroll
    for (int v = 0; v < 2; ++v)
        goff[v] = eoff + (size_t)(32 + 2 * kp[v]) * I_ + n0 + c4[v] * 4;
    int aoff = abase + 64;   // bytes: 32 bf16 per k-step

    constexpr int NSTEP = H_ / 32;
    for (int k = 0; k < NSTEP - 1; ++k) {
        const int cur = k & 1;
        const int nxt = cur ^ 1;
        // issue next-tile loads first (latency hidden behind WMMAs)
        async_ld_b128(aldso[nxt], xb, aoff);
        aoff += 64;
        float4 ga[2], gb[2], ua[2], ub[2];
#pragma unroll
        for (int v = 0; v < 2; ++v) {
            ga[v] = *(const float4*)(Wg + goff[v]);
            gb[v] = *(const float4*)(Wg + goff[v] + I_);
            ua[v] = *(const float4*)(Wu + goff[v]);
            ub[v] = *(const float4*)(Wu + goff[v] + I_);
            goff[v] += 32 * (size_t)I_;
        }
        // compute current tile
        COMPUTE1(cur);
        // finish staging next tile
#pragma unroll
        for (int v = 0; v < 2; ++v) {
            int nn = c4[v] * 4, kk = 2 * kp[v];
            *(unsigned*)&sBg[nxt][nn + 0][kk] = pack_bf16(ga[v].x, gb[v].x);
            *(unsigned*)&sBg[nxt][nn + 1][kk] = pack_bf16(ga[v].y, gb[v].y);
            *(unsigned*)&sBg[nxt][nn + 2][kk] = pack_bf16(ga[v].z, gb[v].z);
            *(unsigned*)&sBg[nxt][nn + 3][kk] = pack_bf16(ga[v].w, gb[v].w);
            *(unsigned*)&sBu[nxt][nn + 0][kk] = pack_bf16(ua[v].x, ub[v].x);
            *(unsigned*)&sBu[nxt][nn + 1][kk] = pack_bf16(ua[v].y, ub[v].y);
            *(unsigned*)&sBu[nxt][nn + 2][kk] = pack_bf16(ua[v].z, ub[v].z);
            *(unsigned*)&sBu[nxt][nn + 3][kk] = pack_bf16(ua[v].w, ub[v].w);
        }
        wait_async0();
        __syncthreads();
    }
    COMPUTE1((NSTEP - 1) & 1);
#undef COMPUTE1

    // epilogue: act = silu(g) * u
    if (m0 + 64 <= ne) {
        // full tile: no per-element predication, packed conversions
#pragma unroll
        for (int ti = 0; ti < 2; ++ti)
#pragma unroll
            for (int tj = 0; tj < 2; ++tj) {
                v8f g = accG[ti * 2 + tj];
                v8f u = accU[ti * 2 + tj];
                const int row0 = es + m0 + wrow * 32 + ti * 16 + kh * 8;
                const int col  = n0 + wcol * 32 + tj * 16 + lm;
                unsigned short* p = act + (size_t)row0 * I_ + col;
#pragma unroll
                for (int j = 0; j < 8; j += 2) {
                    unsigned pk = pack_bf16(silu_fast(g[j]) * u[j],
                                            silu_fast(g[j + 1]) * u[j + 1]);
                    p[(size_t)j * I_]       = (unsigned short)pk;
                    p[(size_t)(j + 1) * I_] = (unsigned short)(pk >> 16);
                }
            }
    } else {
#pragma unroll
        for (int ti = 0; ti < 2; ++ti)
#pragma unroll
            for (int tj = 0; tj < 2; ++tj) {
                v8f g = accG[ti * 2 + tj];
                v8f u = accU[ti * 2 + tj];
#pragma unroll
                for (int j = 0; j < 8; ++j) {
                    int grow = m0 + wrow * 32 + ti * 16 + kh * 8 + j;
                    if (grow < ne) {
                        float a = silu_fast(g[j]) * u[j];
                        int col = n0 + wcol * 32 + tj * 16 + lm;
                        act[(size_t)(es + grow) * I_ + col] = f2bf(a);
                    }
                }
            }
    }
}

// ---------------------------------------------------------------- GEMM 2
// out[tok, :] += w * (act[es+m, :] @ Wd[e]),  f32 atomics
__global__ __launch_bounds__(256) void moe_gemm2(
    const unsigned short* __restrict__ act,
    const float* __restrict__ Wd,
    const int* __restrict__ estart, const int* __restrict__ tokidx,
    const float* __restrict__ tokw, float* __restrict__ out) {
    const int e  = blockIdx.y >> 5;
    const int mt = blockIdx.y & 31;
    const int es = estart[e];
    const int ne = estart[e + 1] - es;
    const int m0 = mt * 64;
    if (m0 >= ne) return;
    const int n0 = blockIdx.x * 128;

    __shared__ __align__(16) unsigned short sA[2][64][40];
    __shared__ __align__(16) unsigned short sB[2][128][40];
    __shared__ float stw[64];
    __shared__ int   stid[64];

    const int tid  = threadIdx.x;
    const int lane = tid & 31;
    const int wv   = tid >> 5;
    const int wrow = wv >> 2;
    const int wcol = wv & 3;
    const int lm   = lane & 15;
    const int kh   = lane >> 4;

    if (tid < 64) {
        int grow = m0 + tid;
        if (grow < ne) { stw[tid] = tokw[es + grow]; stid[tid] = tokidx[es + grow]; }
        else           { stw[tid] = 0.f;             stid[tid] = 0; }
    }

    const size_t eoff = (size_t)e * I_ * H_;

    v8f acc[4];
#pragma unroll
    for (int i = 0; i < 4; ++i) acc[i] = vzero8();

    const int arow = tid >> 2;
    const int aseg = tid & 3;
    const int ar   = (m0 + arow < ne) ? (m0 + arow) : (ne - 1);
    const int abase = (int)((size_t)(es + ar) * I_ * 2) + aseg * 16;
    const unsigned aldso[2] = { lds_off(&sA[0][arow][aseg * 8]),
                                lds_off(&sA[1][arow][aseg * 8]) };

    int kp[2], c4[2];
#pragma unroll
    for (int v = 0; v < 2; ++v) { int idx = v * 256 + tid; kp[v] = idx >> 5; c4[v] = idx & 31; }

#define COMPUTE2(CUR) do {                                                     \
        FragAB a0, a1, b0, b1;                                                 \
        const int m = wrow * 32 + lm;                                          \
        const int n = wcol * 32 + lm;                                          \
        a0.q[0] = *(const uint4*)(&sA[CUR][m][kh * 16]);                       \
        a0.q[1] = *(const uint4*)(&sA[CUR][m][kh * 16 + 8]);                   \
        a1.q[0] = *(const uint4*)(&sA[CUR][m + 16][kh * 16]);                  \
        a1.q[1] = *(const uint4*)(&sA[CUR][m + 16][kh * 16 + 8]);              \
        b0.q[0] = *(const uint4*)(&sB[CUR][n][kh * 16]);                       \
        b0.q[1] = *(const uint4*)(&sB[CUR][n][kh * 16 + 8]);                   \
        b1.q[0] = *(const uint4*)(&sB[CUR][n + 16][kh * 16]);                  \
        b1.q[1] = *(const uint4*)(&sB[CUR][n + 16][kh * 16 + 8]);              \
        acc[0] = wmma_bf16(a0, b0, acc[0]);                                    \
        acc[1] = wmma_bf16(a0, b1, acc[1]);                                    \
        acc[2] = wmma_bf16(a1, b0, acc[2]);                                    \
        acc[3] = wmma_bf16(a1, b1, acc[3]);                                    \
    } while (0)

    // prologue: stage k0 = 0 into buffer 0
    {
        async_ld_b128(aldso[0], act, abase);
#pragma unroll
        for (int v = 0; v < 2; ++v) {
            const size_t o = eoff + (size_t)(2 * kp[v]) * H_ + n0 + c4[v] * 4;
            float4 da = *(const float4*)(Wd + o), db = *(const float4*)(Wd + o + H_);
            int nn = c4[v] * 4, kk = 2 * kp[v];
            *(unsigned*)&sB[0][nn + 0][kk] = pack_bf16(da.x, db.x);
            *(unsigned*)&sB[0][nn + 1][kk] = pack_bf16(da.y, db.y);
            *(unsigned*)&sB[0][nn + 2][kk] = pack_bf16(da.z, db.z);
            *(unsigned*)&sB[0][nn + 3][kk] = pack_bf16(da.w, db.w);
        }
        wait_async0();
        __syncthreads();
    }

    size_t doff[2];
#pragma unroll
    for (int v = 0; v < 2; ++v)
        doff[v] = eoff + (size_t)(32 + 2 * kp[v]) * H_ + n0 + c4[v] * 4;
    int aoff = abase + 64;

    constexpr int NSTEP = I_ / 32;
    for (int k = 0; k < NSTEP - 1; ++k) {
        const int cur = k & 1;
        const int nxt = cur ^ 1;
        async_ld_b128(aldso[nxt], act, aoff);
        aoff += 64;
        float4 da[2], db[2];
#pragma unroll
        for (int v = 0; v < 2; ++v) {
            da[v] = *(const float4*)(Wd + doff[v]);
            db[v] = *(const float4*)(Wd + doff[v] + H_);
            doff[v] += 32 * (size_t)H_;
        }
        COMPUTE2(cur);
#pragma unroll
        for (int v = 0; v < 2; ++v) {
            int nn = c4[v] * 4, kk = 2 * kp[v];
            *(unsigned*)&sB[nxt][nn + 0][kk] = pack_bf16(da[v].x, db[v].x);
            *(unsigned*)&sB[nxt][nn + 1][kk] = pack_bf16(da[v].y, db[v].y);
            *(unsigned*)&sB[nxt][nn + 2][kk] = pack_bf16(da[v].z, db[v].z);
            *(unsigned*)&sB[nxt][nn + 3][kk] = pack_bf16(da[v].w, db[v].w);
        }
        wait_async0();
        __syncthreads();
    }
    COMPUTE2((NSTEP - 1) & 1);
#undef COMPUTE2

    if (m0 + 64 <= ne) {
#pragma unroll
        for (int ti = 0; ti < 2; ++ti)
#pragma unroll
            for (int tj = 0; tj < 2; ++tj) {
                v8f c = acc[ti * 2 + tj];
                const int lrow0 = wrow * 32 + ti * 16 + kh * 8;
                const int col   = n0 + wcol * 32 + tj * 16 + lm;
#pragma unroll
                for (int j = 0; j < 8; ++j) {
                    atomicAdd(&out[(size_t)stid[lrow0 + j] * H_ + col],
                              stw[lrow0 + j] * c[j]);
                }
            }
    } else {
#pragma unroll
        for (int ti = 0; ti < 2; ++ti)
#pragma unroll
            for (int tj = 0; tj < 2; ++tj) {
                v8f c = acc[ti * 2 + tj];
#pragma unroll
                for (int j = 0; j < 8; ++j) {
                    int lrow = wrow * 32 + ti * 16 + kh * 8 + j;
                    int grow = m0 + lrow;
                    if (grow < ne) {
                        int col = n0 + wcol * 32 + tj * 16 + lm;
                        atomicAdd(&out[(size_t)stid[lrow] * H_ + col], stw[lrow] * c[j]);
                    }
                }
            }
    }
}

// ---------------------------------------------------------------- launch
extern "C" void kernel_launch(void* const* d_in, const int* in_sizes, int n_in,
                              void* d_out, int out_size, void* d_ws, size_t ws_size,
                              hipStream_t stream) {
    (void)in_sizes; (void)n_in; (void)out_size; (void)ws_size;
    const float* x  = (const float*)d_in[0];
    const int* ids  = (const int*)d_in[1];
    const float* ew = (const float*)d_in[2];
    const float* Wg = (const float*)d_in[3];
    const float* Wu = (const float*)d_in[4];
    const float* Wd = (const float*)d_in[5];
    float* out = (float*)d_out;

    char* w = (char*)d_ws;
    int*            estart = (int*)(w);                         // 256 B reserved
    int*            tokidx = (int*)(w + 256);                   // B*K ints
    float*          tokw   = (float*)(w + 256 + 8192);          // B*K floats
    unsigned short* xb     = (unsigned short*)(w + 16640);      // B*H bf16
    unsigned short* act    = (unsigned short*)(w + 16640 + (size_t)B_ * H_ * 2);  // B*K*I bf16

    zero_f32<<<dim3((B_ * H_ + 255) / 256), dim3(256), 0, stream>>>(out, B_ * H_);
    cvt_bf16<<<dim3((B_ * H_ / 2 + 255) / 256), dim3(256), 0, stream>>>(x, (unsigned*)xb, B_ * H_ / 2);
    moe_route<<<dim3(1), dim3(256), 0, stream>>>(ids, ew, estart, tokidx, tokw);
    moe_gemm1<<<dim3(I_ / 128, E_ * 32), dim3(256), 0, stream>>>(xb, Wg, Wu, estart, tokidx, act);
    moe_gemm2<<<dim3(H_ / 128, E_ * 32), dim3(256), 0, stream>>>(act, Wd, estart, tokidx, tokw, out);
}
